// MultiHeadSparseAttention_89429809038132
// MI455X (gfx1250) — compile-verified, matrix operand order checked by reading, not run on hardware
//
#include <hip/hip_runtime.h>

typedef __attribute__((ext_vector_type(16))) _Float16 v16h;
typedef __attribute__((ext_vector_type(8)))  float    v8f;

constexpr int Bc = 4;
constexpr int Sc = 8192;
constexpr int Dc = 64;
constexpr int Hc = 8;
constexpr int Lc = 128;
constexpr int NBc = 64;              // number of blocks per sequence
constexpr int HD = Hc * Dc;          // 512
constexpr int Mrows = Bc * Sc;       // 32768

// ---------------------------------------------------------------------------
// WMMA helpers (CDNA5 wave32, 16x16x32 f16 -> f32)
// A fragment (16x32 f16): lane l holds row (l&15); element e holds
//   k = (e<8 ? e : e+8) + ((l>=16) ? 8 : 0)   -> two contiguous 8-half chunks.
// B fragment for D = A * B with B[k][n] from a row-major source holding B^T
//   rows (source[n][k]): identical addressing with row = n.
// C/D fragment (16x16 f32): element e -> row = e + 8*(lane>=16), col = lane&15.
// ---------------------------------------------------------------------------

static __device__ __forceinline__ v8f wmma_f16(v16h a, v16h b, v8f c) {
  return __builtin_amdgcn_wmma_f32_16x16x32_f16(false, a, false, b, (short)0, c,
                                                false, false);
}

static __device__ __forceinline__ v16h frag_ld_h(const _Float16* p, int ld) {
  const int lane = threadIdx.x & 31;
  const _Float16* q = p + (size_t)(lane & 15) * ld + ((lane >> 4) << 3);
  v16h f;
  *reinterpret_cast<uint4*>(&f)       = *reinterpret_cast<const uint4*>(q);
  *(reinterpret_cast<uint4*>(&f) + 1) = *reinterpret_cast<const uint4*>(q + 16);
  return f;
}

static __device__ __forceinline__ v16h frag_ld_f32(const float* p, int ld) {
  const int lane = threadIdx.x & 31;
  const float* q = p + (size_t)(lane & 15) * ld + ((lane >> 4) << 3);
  v16h f;
#pragma unroll
  for (int i = 0; i < 8; ++i) f[i] = (_Float16)q[i];
#pragma unroll
  for (int i = 0; i < 8; ++i) f[i + 8] = (_Float16)q[i + 16];
  return f;
}

// Async bulk copy global -> LDS, 16 bytes per lane (ASYNCcnt tracked).
// Generic LDS pointer low 32 bits == LDS byte offset (ISA 10.2 aperture).
static __device__ __forceinline__ void async_ld16(uint32_t lds_off,
                                                  const void* gaddr) {
  asm volatile("global_load_async_to_lds_b128 %0, %1, off"
               :
               : "v"(lds_off), "v"(gaddr)
               : "memory");
}

static __device__ __forceinline__ void wait_async0() {
  asm volatile("s_wait_asynccnt 0x0" ::: "memory");
}

// ---------------------------------------------------------------------------
// Kernel 1: fused Q/K/V projection. y = x @ W^T + b, reshaped to [B,n,H,L,D].
// rQ / rK stored row-major (L x D) per (b,blk,h); rV stored transposed (D x L)
// via an LDS tile transpose so global stores are coalesced b128.
// grid = (Mrows/16, 3), block = 256 (8 waves = 8 heads).
// ---------------------------------------------------------------------------
__global__ void __launch_bounds__(256)
proj_kernel(const float* __restrict__ Qp, const float* __restrict__ Kp,
            const float* __restrict__ Vp,
            const float* __restrict__ WQ, const float* __restrict__ bQ,
            const float* __restrict__ WK, const float* __restrict__ bK,
            const float* __restrict__ WV, const float* __restrict__ bV,
            _Float16* __restrict__ rQ, _Float16* __restrict__ rK,
            _Float16* __restrict__ rVt) {
  __shared__ __align__(16) _Float16 sT[HD * 16];  // 16 KB transpose staging

  const int which = blockIdx.y;
  const float* X    = which == 0 ? Qp : (which == 1 ? Kp : Vp);
  const float* W    = which == 0 ? WQ : (which == 1 ? WK : WV);
  const float* bias = which == 0 ? bQ : (which == 1 ? bK : bV);

  const int m0 = blockIdx.x * 16;              // global row into [Mrows, 64]
  const int h  = threadIdx.x >> 5;             // wave = head
  const int lane = threadIdx.x & 31;
  const int lane15 = lane & 15;
  const int hi8 = (lane >> 4) << 3;

  const v16h a0 = frag_ld_f32(X + (size_t)m0 * Dc, Dc);       // k 0..31
  const v16h a1 = frag_ld_f32(X + (size_t)m0 * Dc + 32, Dc);  // k 32..63

  const int b    = m0 >> 13;
  const int srem = m0 & (Sc - 1);
  const int nb   = srem >> 7;
  const int l0   = srem & (Lc - 1);
  const size_t obase = ((size_t)(b * NBc + nb) * Hc + h) * (Lc * Dc);

#pragma unroll
  for (int nt = 0; nt < 4; ++nt) {
    const int c0 = h * 64 + nt * 16;
    v8f c = {};
    c = wmma_f16(a0, frag_ld_f32(W + (size_t)c0 * Dc, Dc), c);
    c = wmma_f16(a1, frag_ld_f32(W + (size_t)c0 * Dc + 32, Dc), c);
    const float bv = bias[c0 + lane15];
    if (which < 2) {
      _Float16* dst = (which == 0 ? rQ : rK) + obase + (size_t)l0 * Dc +
                      nt * 16 + lane15;
#pragma unroll
      for (int e = 0; e < 8; ++e)
        dst[(size_t)(e + hi8) * Dc] = (_Float16)(c[e] + bv);
    } else {
      // transpose through LDS: sT[c][l_local]
#pragma unroll
      for (int e = 0; e < 8; ++e)
        sT[(c0 + lane15) * 16 + e + hi8] = (_Float16)(c[e] + bv);
    }
  }

  if (which == 2) {
    __syncthreads();
    // coalesced store: each of 512 columns c becomes a 32-byte run of rVt.
    // thread t handles columns 2t and 2t+1.
#pragma unroll
    for (int r = 0; r < 2; ++r) {
      const int c = threadIdx.x * 2 + r;
      const int h2 = c >> 6, d = c & 63;
      _Float16* dst = rVt + ((size_t)(b * NBc + nb) * Hc + h2) * (Lc * Dc) +
                      (size_t)d * Lc + l0;
      const uint4* src = (const uint4*)(sT + c * 16);
      ((uint4*)dst)[0] = src[0];
      ((uint4*)dst)[1] = src[1];
    }
  }
}

// ---------------------------------------------------------------------------
// Kernel 2: block-local causal attention, one workgroup per (b, blk, h).
// K and V^T staged into LDS with async global->LDS b128 loads (ASYNCcnt).
// 8 waves; wave w owns rows [16w, 16w+16) of the 128x128 score strip.
// Writes P-weighted V directly into merged [B,S,H*D] f16 layout and
// atomically accumulates the block pooling sum.
// ---------------------------------------------------------------------------
__global__ void __launch_bounds__(256)
attn_block_kernel(const _Float16* __restrict__ rQ,
                  const _Float16* __restrict__ rK,
                  const _Float16* __restrict__ rVt,
                  _Float16* __restrict__ merged, float* __restrict__ pool) {
  __shared__ __align__(16) _Float16 sK[Lc * Dc];        // 16 KB
  __shared__ __align__(16) _Float16 sVt[Dc * Lc];       // 16 KB
  __shared__ __align__(16) _Float16 sP[8 * 16 * Lc];    // 32 KB

  const int blk = blockIdx.x;            // b*NB*H
  const int h  = blk % Hc;
  const int nb = (blk / Hc) % NBc;
  const int b  = blk / (Hc * NBc);
  const size_t base = ((size_t)(b * NBc + nb) * Hc + h) * (Lc * Dc);

  {  // async stage of K and V^T into LDS, 16B per lane per op
    const uint32_t ldsK = (uint32_t)(uintptr_t)(void*)sK;
    const uint32_t ldsV = (uint32_t)(uintptr_t)(void*)sVt;
    const _Float16* gK = rK + base;
    const _Float16* gV = rVt + base;
    for (int i = threadIdx.x; i < (Lc * Dc) / 8; i += 256) {
      async_ld16(ldsK + i * 16, gK + i * 8);
      async_ld16(ldsV + i * 16, gV + i * 8);
    }
    wait_async0();
  }
  __syncthreads();

  const int w = threadIdx.x >> 5;
  const int lane = threadIdx.x & 31;
  const int lane15 = lane & 15;
  const int hi8 = (lane >> 4) << 3;

  const v16h qf0 = frag_ld_h(rQ + base + (size_t)(w * 16) * Dc, Dc);
  const v16h qf1 = frag_ld_h(rQ + base + (size_t)(w * 16) * Dc + 32, Dc);

  v8f acc[8];
#pragma unroll
  for (int nt = 0; nt < 8; ++nt) {
    v8f c = {};
    c = wmma_f16(qf0, frag_ld_h(sK + nt * 16 * Dc, Dc), c);
    c = wmma_f16(qf1, frag_ld_h(sK + nt * 16 * Dc + 32, Dc), c);
    acc[nt] = c;
  }

  // scale + causal mask + row softmax (rows live in elem e across the 16-lane
  // half; reduce over nt then over lanes with xor 1,2,4,8).
  const float scale = 0.125f;  // 1/sqrt(64)
  float rmax[8], rsum[8];
#pragma unroll
  for (int e = 0; e < 8; ++e) rmax[e] = -3.0e38f;
#pragma unroll
  for (int nt = 0; nt < 8; ++nt) {
#pragma unroll
    for (int e = 0; e < 8; ++e) {
      const int row = w * 16 + e + hi8;
      const int col = nt * 16 + lane15;
      float s = acc[nt][e] * scale + (col > row ? -1.0e9f : 0.0f);
      acc[nt][e] = s;
      rmax[e] = fmaxf(rmax[e], s);
    }
  }
#pragma unroll
  for (int e = 0; e < 8; ++e) {
#pragma unroll
    for (int m = 1; m <= 8; m <<= 1)
      rmax[e] = fmaxf(rmax[e], __shfl_xor(rmax[e], m, 32));
    rsum[e] = 0.0f;
  }
#pragma unroll
  for (int nt = 0; nt < 8; ++nt)
#pragma unroll
    for (int e = 0; e < 8; ++e) {
      const float p = __expf(acc[nt][e] - rmax[e]);
      acc[nt][e] = p;
      rsum[e] += p;
    }
#pragma unroll
  for (int e = 0; e < 8; ++e) {
#pragma unroll
    for (int m = 1; m <= 8; m <<= 1) rsum[e] += __shfl_xor(rsum[e], m, 32);
    rsum[e] = 1.0f / rsum[e];
  }

  // P strip to wave-private LDS (C-fragment -> row-major 16x128 f16)
  _Float16* myP = sP + w * 16 * Lc;
#pragma unroll
  for (int nt = 0; nt < 8; ++nt)
#pragma unroll
    for (int e = 0; e < 8; ++e)
      myP[(e + hi8) * Lc + nt * 16 + lane15] =
          (_Float16)(acc[nt][e] * rsum[e]);

  // O = P (16x128) x V (128x64); B-fragments come from V^T rows (contiguous)
  v16h pa[4];
#pragma unroll
  for (int kc = 0; kc < 4; ++kc) pa[kc] = frag_ld_h(myP + kc * 32, Lc);

  const size_t srow = (size_t)(b * NBc + nb) * Lc + w * 16;
#pragma unroll
  for (int dt = 0; dt < 4; ++dt) {
    v8f o = {};
#pragma unroll
    for (int kc = 0; kc < 4; ++kc)
      o = wmma_f16(pa[kc], frag_ld_h(sVt + dt * 16 * Lc + kc * 32, Lc), o);
    // store into merged [B,S,512] (s = nb*L + l, c = h*64 + d)
#pragma unroll
    for (int e = 0; e < 8; ++e)
      merged[(srow + e + hi8) * (size_t)HD + h * 64 + dt * 16 + lane15] =
          (_Float16)o[e];
    // pooling partial: column sums over this wave's 16 rows
    float cs = 0.0f;
#pragma unroll
    for (int e = 0; e < 8; ++e) cs += o[e];
    cs += __shfl_xor(cs, 16, 32);
    if (lane < 16)
      atomicAdd(pool + ((size_t)(b * Hc + h) * NBc + nb) * Dc + dt * 16 + lane,
                cs);
  }
}

// ---------------------------------------------------------------------------
// Kernel 3: pooled global attention per (b,h): 64x64 scores over block means.
// Tiny (~34 MFLOP total) -> plain f32 VALU.
// ---------------------------------------------------------------------------
__global__ void __launch_bounds__(64)
global_attn_kernel(const float* __restrict__ pool, float* __restrict__ g) {
  __shared__ float sPg[NBc * Dc];   // 16 KB
  __shared__ float sS[NBc * 64];    // 16 KB, sS[k*64 + q]
  const int bh = blockIdx.x;
  const float* src = pool + (size_t)bh * NBc * Dc;
  for (int i = threadIdx.x; i < NBc * Dc; i += 64)
    sPg[i] = src[i] * (1.0f / 128.0f);   // mean over L
  __syncthreads();

  const int q = threadIdx.x;
  float m = -3.0e38f;
  for (int k = 0; k < NBc; ++k) {
    float dot = 0.0f;
#pragma unroll 8
    for (int d = 0; d < Dc; ++d) dot += sPg[q * Dc + d] * sPg[k * Dc + d];
    dot = dot * 0.125f + (k > q ? -1.0e9f : 0.0f);
    sS[k * 64 + q] = dot;
    m = fmaxf(m, dot);
  }
  float sum = 0.0f;
  for (int k = 0; k < NBc; ++k) {
    const float p = __expf(sS[k * 64 + q] - m);
    sS[k * 64 + q] = p;
    sum += p;
  }
  const float inv = 1.0f / sum;
  float* dst = g + (size_t)bh * NBc * Dc + (size_t)q * Dc;
  for (int d = 0; d < Dc; ++d) {
    float o = 0.0f;
#pragma unroll 8
    for (int k = 0; k < NBc; ++k) o += sS[k * 64 + q] * sPg[k * Dc + d];
    dst[d] = o * inv;
  }
}

// ---------------------------------------------------------------------------
// Kernel 4: broadcast-add global context into merged f16 (8-wide vectorized).
// ---------------------------------------------------------------------------
__global__ void __launch_bounds__(256)
merge_add_kernel(_Float16* __restrict__ merged, const float* __restrict__ g) {
  const size_t i8 = ((size_t)blockIdx.x * 256 + threadIdx.x) * 8;  // B*S*512
  const int c = (int)(i8 & (HD - 1));
  const size_t srow = i8 >> 9;
  const int h = c >> 6, d0 = c & 63;
  const int b = (int)(srow >> 13);
  const int nb = (int)((srow >> 7) & (NBc - 1));
  const float* gp = g + ((size_t)(b * Hc + h) * NBc + nb) * Dc + d0;
  _Float16* mp = merged + i8;
  uint4 raw = *(const uint4*)mp;
  _Float16* hp = (_Float16*)&raw;
#pragma unroll
  for (int j = 0; j < 8; ++j) hp[j] = (_Float16)((float)hp[j] + gp[j]);
  *(uint4*)mp = raw;
}

// ---------------------------------------------------------------------------
// Kernel 5: output projection: out = merged[32768,512] @ W_w^T[512,64] + b.
// Dual accumulators for WMMA ILP; fp32 output.
// ---------------------------------------------------------------------------
__global__ void __launch_bounds__(256)
outproj_kernel(const _Float16* __restrict__ merged,
               const float* __restrict__ Ww, const float* __restrict__ Wb,
               float* __restrict__ out) {
  const int w = threadIdx.x >> 5;
  const int lane = threadIdx.x & 31;
  const int lane15 = lane & 15;
  const int hi8 = (lane >> 4) << 3;
  const int m0 = blockIdx.x * 32 + (w >> 2) * 16;
  const int c0 = (w & 3) * 16;

  v8f acc0 = {}, acc1 = {};
#pragma unroll
  for (int kc = 0; kc < 16; kc += 2) {
    const v16h a0 = frag_ld_h(merged + (size_t)m0 * HD + kc * 32, HD);
    const v16h b0 = frag_ld_f32(Ww + (size_t)c0 * HD + kc * 32, HD);
    acc0 = wmma_f16(a0, b0, acc0);
    const v16h a1 = frag_ld_h(merged + (size_t)m0 * HD + (kc + 1) * 32, HD);
    const v16h b1 = frag_ld_f32(Ww + (size_t)c0 * HD + (kc + 1) * 32, HD);
    acc1 = wmma_f16(a1, b1, acc1);
  }
  const float bv = Wb[c0 + lane15];
#pragma unroll
  for (int e = 0; e < 8; ++e)
    out[(size_t)(m0 + e + hi8) * Dc + c0 + lane15] = acc0[e] + acc1[e] + bv;
}

// ---------------------------------------------------------------------------
extern "C" void kernel_launch(void* const* d_in, const int* in_sizes, int n_in,
                              void* d_out, int out_size, void* d_ws,
                              size_t ws_size, hipStream_t stream) {
  const float* Q    = (const float*)d_in[0];
  const float* K    = (const float*)d_in[1];
  const float* V    = (const float*)d_in[2];
  const float* WQ_w = (const float*)d_in[3];
  const float* WQ_b = (const float*)d_in[4];
  const float* WK_w = (const float*)d_in[5];
  const float* WK_b = (const float*)d_in[6];
  const float* WV_w = (const float*)d_in[7];
  const float* WV_b = (const float*)d_in[8];
  const float* W_w  = (const float*)d_in[9];
  const float* W_b  = (const float*)d_in[10];

  char* ws = (char*)d_ws;
  const size_t halfMat = (size_t)Mrows * HD * sizeof(_Float16);  // 32 MB
  _Float16* rQ     = (_Float16*)(ws);
  _Float16* rK     = (_Float16*)(ws + halfMat);
  _Float16* rVt    = (_Float16*)(ws + 2 * halfMat);
  _Float16* merged = (_Float16*)(ws + 3 * halfMat);
  float* pool = (float*)(ws + 4 * halfMat);                       // 512 KB
  float* g    = (float*)(ws + 4 * halfMat + (size_t)Bc * Hc * NBc * Dc * 4);

  hipMemsetAsync(pool, 0, (size_t)Bc * Hc * NBc * Dc * sizeof(float), stream);

  proj_kernel<<<dim3(Mrows / 16, 3), 256, 0, stream>>>(
      Q, K, V, WQ_w, WQ_b, WK_w, WK_b, WV_w, WV_b, rQ, rK, rVt);

  attn_block_kernel<<<Bc * NBc * Hc, 256, 0, stream>>>(rQ, rK, rVt, merged,
                                                       pool);

  global_attn_kernel<<<Bc * Hc, 64, 0, stream>>>(pool, g);

  merge_add_kernel<<<(Mrows * HD) / (256 * 8), 256, 0, stream>>>(merged, g);

  outproj_kernel<<<Mrows / 32, 256, 0, stream>>>(merged, W_w, W_b,
                                                 (float*)d_out);
}
